// MoEACTBlock_62225486185201
// MI455X (gfx1250) — matrix operand order repaired and checked by reading, not compiled, not run on hardware
//
#include <hip/hip_runtime.h>
#include <hip/hip_bf16.h>

typedef __bf16 bf16;
typedef __attribute__((ext_vector_type(16))) __bf16 v16bf;
typedef __attribute__((ext_vector_type(8)))  __bf16 v8bf;
typedef __attribute__((ext_vector_type(8)))  float  v8f;
typedef __attribute__((ext_vector_type(4)))  unsigned int uint32x4;
typedef __attribute__((ext_vector_type(8)))  int int32x8;
typedef __attribute__((ext_vector_type(4)))  int int32x4;

#define TOK   8192   // B*S
#define HDIM  1024
#define NEXP  8
#define TOPK  2
#define FDIM  4096
#define LN_EPS 1e-5f

#define TM 128
#define TN 128
#define TK 32
#define LDSP (TK + 8)                 // row pad: 40 bf16 = 80B (20 DW) -> conflict-free
#define A_BYTES (TM * LDSP * 2)       // 10240 B per A tile
#define STAGE_BYTES (2 * A_BYTES)     // A tile + B tile = 20480 B per stage
#define SMEM_BYTES (2 * STAGE_BYTES)  // double buffered = 40960 B

#if __has_builtin(__builtin_amdgcn_tensor_load_to_lds)
#define USE_TDM 1
#else
#define USE_TDM 0
#endif

// ---------------------------------------------------------------- helpers
__device__ __forceinline__ float gelu_tanh(float x) {
    float x3 = x * x * x;
    return 0.5f * x * (1.0f + tanhf(0.7978845608028654f * (x + 0.044715f * x3)));
}

__device__ __forceinline__ v16bf mk16(v8bf lo, v8bf hi) {
    v16bf r;
#pragma unroll
    for (int i = 0; i < 8; ++i) { r[i] = lo[i]; r[i + 8] = hi[i]; }
    return r;
}

__device__ __forceinline__ float wave_sum(float v) {
#pragma unroll
    for (int o = 16; o > 0; o >>= 1) v += __shfl_xor(v, o, 32);
    return v;
}

// CDNA5 async copy: global -> LDS, 16B per lane, tracked by ASYNCcnt.
// lds_off is the byte offset inside the workgroup LDS allocation (dynamic LDS
// starts at offset 0 because these kernels declare no static LDS).
__device__ __forceinline__ void async_ld16(unsigned lds_off, const void* gaddr) {
    asm volatile("global_load_async_to_lds_b128 %0, %1, off"
                 :: "v"(lds_off), "v"(gaddr) : "memory");
}
#if USE_TDM
#define WAIT_ASYNC_PIPE() asm volatile("s_wait_asynccnt 0x2" ::: "memory")
#else
#define WAIT_ASYNC_PIPE() asm volatile("s_wait_asynccnt 0x4" ::: "memory")
#endif
#define WAIT_ASYNC_LAST() asm volatile("s_wait_asynccnt 0x0" ::: "memory")

#if USE_TDM
// Tensor Data Mover: DMA a [TM rows x TK bf16] tile of a [rows x lda] bf16
// tensor into LDS at byte offset lds_off, with LDS padding 16DW data + 4DW pad
// per row => 80B pitch (matches LDSP).  D# layout per CDNA5 ISA 8.3/8.4.
// This toolchain's builtin takes 6 args: (g0, g1, g2, g3, g4, cpol).
__device__ __forceinline__ void tdm_load_tile(const bf16* gptr, unsigned lds_off,
                                              int rows, int lda) {
    unsigned long long ga = (unsigned long long)gptr;
    uint32x4 g0;
    g0[0] = 1u;                                            // count=1, no gather
    g0[1] = lds_off;                                       // lds_addr (bytes)
    g0[2] = (unsigned)(ga & 0xFFFFFFFFu);                  // global_addr[31:0]
    g0[3] = (unsigned)((ga >> 32) & 0x1FFFFFFu) | (2u << 30); // addr[56:32]|type=2
    int32x8 g1;
    g1[0] = (1 << 16) | (1 << 20) | (3 << 22) | (3 << 25); // 2B elems, pad 16DW/4DW
    g1[1] = (int)(((unsigned)lda & 0xFFFFu) << 16);        // tensor_dim0 lo16
    g1[2] = (int)((((unsigned)lda >> 16) & 0xFFFFu) |
                  (((unsigned)rows & 0xFFFFu) << 16));     // dim0 hi16 | dim1 lo16
    g1[3] = (int)((((unsigned)rows >> 16) & 0xFFFFu) |
                  ((unsigned)TK << 16));                   // dim1 hi16 | tile_dim0
    g1[4] = TM;                                            // tile_dim1 (tile_dim2=0)
    g1[5] = lda;                                           // dim0_stride lo32
    g1[6] = 0; g1[7] = 0;
    int32x4 g2 = {0, 0, 0, 0};
    int32x4 g3 = {0, 0, 0, 0};
    int32x8 g4 = {0, 0, 0, 0, 0, 0, 0, 0};
    __builtin_amdgcn_tensor_load_to_lds(g0, g1, g2, g3, g4, 0);
}
#endif

// ------------------------------------------------- router + LN-stats + xhat
__global__ __launch_bounds__(256) void router_ln_kernel(
    const float* __restrict__ x, const float* __restrict__ rw,
    const float* __restrict__ rb, bf16* __restrict__ xhat,
    int* __restrict__ top_e, float* __restrict__ top_w, int* __restrict__ cnt)
{
    int t = blockIdx.x;
    int tid = threadIdx.x;
    const float* xr = x + (size_t)t * HDIM;
    float xv[4];
    float s = 0.f, q = 0.f, dot[NEXP];
#pragma unroll
    for (int e = 0; e < NEXP; ++e) dot[e] = 0.f;
#pragma unroll
    for (int i = 0; i < 4; ++i) {
        int h = tid + i * 256;
        float v = xr[h];
        xv[i] = v; s += v; q += v * v;
        const float* r = rw + (size_t)h * NEXP;
#pragma unroll
        for (int e = 0; e < NEXP; ++e) dot[e] += v * r[e];
    }
    s = wave_sum(s); q = wave_sum(q);
#pragma unroll
    for (int e = 0; e < NEXP; ++e) dot[e] = wave_sum(dot[e]);

    __shared__ float red[8][NEXP + 2];
    __shared__ float bc[2];
    int wave = tid >> 5, lane = tid & 31;
    if (lane == 0) {
        red[wave][0] = s; red[wave][1] = q;
#pragma unroll
        for (int e = 0; e < NEXP; ++e) red[wave][2 + e] = dot[e];
    }
    __syncthreads();
    if (tid == 0) {
        float ss = 0.f, qq = 0.f, d[NEXP];
#pragma unroll
        for (int e = 0; e < NEXP; ++e) d[e] = 0.f;
        for (int w = 0; w < 8; ++w) {
            ss += red[w][0]; qq += red[w][1];
#pragma unroll
            for (int e = 0; e < NEXP; ++e) d[e] += red[w][2 + e];
        }
        float mu = ss * (1.0f / HDIM);
        float var = qq * (1.0f / HDIM) - mu * mu;
        float rstd = rsqrtf(var + LN_EPS);
        bc[0] = mu; bc[1] = rstd;
        float lg[NEXP], mx = -1e30f;
#pragma unroll
        for (int e = 0; e < NEXP; ++e) { lg[e] = d[e] + rb[e]; mx = fmaxf(mx, lg[e]); }
        float p[NEXP], ps = 0.f;
#pragma unroll
        for (int e = 0; e < NEXP; ++e) { p[e] = __expf(lg[e] - mx); ps += p[e]; }
        float inv = 1.0f / ps;
#pragma unroll
        for (int e = 0; e < NEXP; ++e) p[e] *= inv;
        int i0 = 0;
#pragma unroll
        for (int e = 1; e < NEXP; ++e) if (p[e] > p[i0]) i0 = e;
        int i1 = (i0 == 0) ? 1 : 0;
#pragma unroll
        for (int e = 0; e < NEXP; ++e) if (e != i0 && p[e] > p[i1]) i1 = e;
        float wsum = p[i0] + p[i1];
        top_e[t * 2 + 0] = i0; top_e[t * 2 + 1] = i1;
        top_w[t * 2 + 0] = p[i0] / wsum; top_w[t * 2 + 1] = p[i1] / wsum;
        atomicAdd(&cnt[i0], 1); atomicAdd(&cnt[i1], 1);
    }
    __syncthreads();
    float mu = bc[0], rstd = bc[1];
#pragma unroll
    for (int i = 0; i < 4; ++i) {
        int h = tid + i * 256;
        xhat[(size_t)t * HDIM + h] = (bf16)((xv[i] - mu) * rstd);
    }
}

__global__ void zero_cnt_kernel(int* __restrict__ cnt) {
    if (threadIdx.x < NEXP) cnt[threadIdx.x] = 0;
}

__global__ void scan_kernel(const int* __restrict__ cnt, int* __restrict__ segoff,
                            int* __restrict__ fill) {
    if (threadIdx.x == 0) {
        int acc = 0;
        for (int e = 0; e < NEXP; ++e) { segoff[e] = acc; acc += cnt[e]; fill[e] = 0; }
        segoff[NEXP] = acc;
    }
}

__global__ __launch_bounds__(256) void assign_kernel(
    const int* __restrict__ top_e, const int* __restrict__ segoff,
    int* __restrict__ fill, int* __restrict__ idx, int* __restrict__ slot)
{
    int t = blockIdx.x * 256 + threadIdx.x;
    if (t >= TOK) return;
#pragma unroll
    for (int k = 0; k < TOPK; ++k) {
        int e = top_e[t * 2 + k];
        int pos = atomicAdd(&fill[e], 1);
        int s = segoff[e] + pos;
        idx[s] = t;
        slot[t * 2 + k] = s;
    }
}

// -------------------------------------- weight convert / transpose kernels
// w1t[e][f][h] = bf16(w1[e][h][f] * ln_g[e][h])
__global__ __launch_bounds__(256) void tr_w1_kernel(
    const float* __restrict__ w1, const float* __restrict__ lng, bf16* __restrict__ w1t)
{
    int e = blockIdx.z;
    int f0 = blockIdx.x * 32;
    int h0 = blockIdx.y * 32;
    __shared__ bf16 tile[32][33];
    int c = threadIdx.x & 31;
    int r = threadIdx.x >> 5;
#pragma unroll
    for (int i = 0; i < 4; ++i) {
        int hh = r + i * 8;
        float g = lng[(size_t)e * HDIM + h0 + hh];
        tile[hh][c] = (bf16)(w1[((size_t)e * HDIM + h0 + hh) * FDIM + f0 + c] * g);
    }
    __syncthreads();
#pragma unroll
    for (int i = 0; i < 4; ++i) {
        int ff = r + i * 8;
        w1t[((size_t)e * FDIM + f0 + ff) * HDIM + h0 + c] = tile[c][ff];
    }
}

// w2t[e][h][f] = bf16(w2[e][f][h])
__global__ __launch_bounds__(256) void tr_w2_kernel(
    const float* __restrict__ w2, bf16* __restrict__ w2t)
{
    int e = blockIdx.z;
    int h0 = blockIdx.x * 32;
    int f0 = blockIdx.y * 32;
    __shared__ bf16 tile[32][33];
    int c = threadIdx.x & 31;
    int r = threadIdx.x >> 5;
#pragma unroll
    for (int i = 0; i < 4; ++i) {
        int ff = r + i * 8;
        tile[ff][c] = (bf16)(w2[((size_t)e * FDIM + f0 + ff) * HDIM + h0 + c]);
    }
    __syncthreads();
#pragma unroll
    for (int i = 0; i < 4; ++i) {
        int hh = r + i * 8;
        w2t[((size_t)e * HDIM + h0 + hh) * FDIM + f0 + c] = tile[c][hh];
    }
}

// b1p[e][f] = b1[e][f] + sum_h ln_b[e][h] * w1[e][h][f]
__global__ __launch_bounds__(256) void foldb1_kernel(
    const float* __restrict__ w1, const float* __restrict__ lnb,
    const float* __restrict__ b1, float* __restrict__ b1p)
{
    int e = blockIdx.y;
    int f = blockIdx.x * 256 + threadIdx.x;
    float acc = b1[(size_t)e * FDIM + f];
    const float* wcol = w1 + (size_t)e * HDIM * FDIM + f;
    const float* lb = lnb + (size_t)e * HDIM;
    for (int h = 0; h < HDIM; ++h)
        acc += lb[h] * wcol[(size_t)h * FDIM];
    b1p[(size_t)e * FDIM + f] = acc;
}

// -------------------------------------------------- grouped WMMA GEMM core
// Double-buffered pipeline:
//   A tile: per-lane GLOBAL_LOAD_ASYNC_TO_LDS_B128 (gathered token rows)
//   B tile: one TENSOR_LOAD_TO_LDS DMA per stage (wave 0), TENSORcnt-tracked
// ASYNCcnt/TENSORcnt complete in order, so waiting <=2 / <=1 guarantees the
// previous stage landed; the workgroup barrier then publishes all portions.
// GELU_EP=1: C = gelu(A@B + bias) -> bf16 outb   (A rows gathered via idx)
// GELU_EP=0: C = A@B              -> f32  outf   (A rows direct from segment)
template <int GELU_EP>
__global__ __launch_bounds__(256) void moe_gemm_kernel(
    const bf16* __restrict__ Abase, int lda,           // K dim
    const bf16* __restrict__ Bmat, int Ntot,           // Bmat: [NEXP][Ntot][lda]
    const float* __restrict__ bias,                    // [NEXP][Ntot] or null
    const int* __restrict__ idx,                       // gather list or null
    const int* __restrict__ segoff,                    // [NEXP+1]
    bf16* __restrict__ outb, float* __restrict__ outf)
{
    extern __shared__ char smem[];   // no static LDS in this kernel -> offset 0

    int e = blockIdx.z;
    int base = segoff[e];
    int cnt = segoff[e + 1] - base;
    int m0 = blockIdx.y * TM;
    if (m0 >= cnt) return;                 // block-uniform: EXEC stays all-ones
    int n0 = blockIdx.x * TN;

    int tid = threadIdx.x;
    int wave = tid >> 5;
    int lane = tid & 31;
    int wm = wave & 3;                     // 4 waves along M (x32)
    int wn = wave >> 2;                    // 2 waves along N (x64)
    int half = lane >> 4;
    int l = lane & 15;

    // A staging: 128 rows x 32 bf16; 2 threads/row, 16 bf16 (32B) each
    int lr = tid >> 1;
    int lc = (tid & 1) * 16;
    unsigned aoff = (unsigned)(lr * (LDSP * 2) + lc * 2);            // A tile byte off
#if !USE_TDM
    unsigned boff = (unsigned)(A_BYTES + lr * (LDSP * 2) + lc * 2);  // B tile byte off
#endif

    int mrow = m0 + lr; if (mrow >= cnt) mrow = cnt - 1;
    size_t arow = idx ? (size_t)idx[base + mrow] : (size_t)(base + mrow);
    const bf16* aptr = Abase + arow * lda + lc;
    const bf16* btile = Bmat + ((size_t)e * Ntot + n0) * lda;  // tile row 0
#if !USE_TDM
    const bf16* bptr = btile + (size_t)lr * lda + lc;
#endif

    v8f acc[2][4];
#pragma unroll
    for (int a = 0; a < 2; ++a)
#pragma unroll
        for (int b = 0; b < 4; ++b)
#pragma unroll
            for (int j = 0; j < 8; ++j) acc[a][b][j] = 0.f;

    int nk = lda / TK;

    // prologue: stage 0 in flight
    async_ld16(aoff,      aptr);
    async_ld16(aoff + 16, aptr + 8);
#if USE_TDM
    if (wave == 0) tdm_load_tile(btile, A_BYTES, Ntot, lda);
#else
    async_ld16(boff,      bptr);
    async_ld16(boff + 16, bptr + 8);
#endif

    for (int kt = 0; kt < nk; ++kt) {
        if (kt + 1 < nk) {
            unsigned sb = (unsigned)(((kt + 1) & 1) * STAGE_BYTES);
            const bf16* ag = aptr + (size_t)(kt + 1) * TK;
            async_ld16(sb + aoff,      ag);
            async_ld16(sb + aoff + 16, ag + 8);
#if USE_TDM
            if (wave == 0)
                tdm_load_tile(btile + (size_t)(kt + 1) * TK, sb + A_BYTES, Ntot, lda);
#else
            const bf16* bg = bptr + (size_t)(kt + 1) * TK;
            async_ld16(sb + boff,      bg);
            async_ld16(sb + boff + 16, bg + 8);
#endif
            WAIT_ASYNC_PIPE();      // stage kt's A transfers have landed
#if USE_TDM
            if (wave == 0) __builtin_amdgcn_s_wait_tensorcnt((short)1);
#endif
        } else {
            WAIT_ASYNC_LAST();
#if USE_TDM
            if (wave == 0) __builtin_amdgcn_s_wait_tensorcnt((short)0);
#endif
        }
        __syncthreads();            // all waves' stage-kt data visible

        const bf16* As = (const bf16*)(smem + (kt & 1) * STAGE_BYTES);
        const bf16* Bs = (const bf16*)(smem + (kt & 1) * STAGE_BYTES + A_BYTES);

        v16bf af[2], bfr[4];
#pragma unroll
        for (int tf = 0; tf < 2; ++tf) {
            int r = wm * 32 + tf * 16 + l;   // A: half -> K {0..7,16..23}/{8..15,24..31}
            af[tf] = mk16(*(const v8bf*)&As[r * LDSP + half * 8],
                          *(const v8bf*)&As[r * LDSP + 16 + half * 8]);
        }
#pragma unroll
        for (int nf = 0; nf < 4; ++nf) {
            int n = wn * 64 + nf * 16 + l;   // B: half -> K {0..7,16..23}/{8..15,24..31}
            const bf16* p = &Bs[n * LDSP + half * 8];
            bfr[nf] = mk16(*(const v8bf*)p, *(const v8bf*)(p + 16));
        }
#pragma unroll
        for (int tf = 0; tf < 2; ++tf)
#pragma unroll
            for (int nf = 0; nf < 4; ++nf)
                acc[tf][nf] = __builtin_amdgcn_wmma_f32_16x16x32_bf16(
                    false, af[tf], false, bfr[nf], (short)0, acc[tf][nf],
                    false, false);

        __syncthreads();            // stage buffer free for reuse at kt+2
    }

    // epilogue: C layout -> M = tf*16 + half*8 + j, N = nf*16 + l
#pragma unroll
    for (int tf = 0; tf < 2; ++tf) {
#pragma unroll
        for (int nf = 0; nf < 4; ++nf) {
            int n = n0 + wn * 64 + nf * 16 + l;
            int mb = m0 + wm * 32 + tf * 16 + half * 8;
            if (GELU_EP) {
                float bia = bias[(size_t)e * Ntot + n];
#pragma unroll
                for (int j = 0; j < 8; ++j) {
                    int m = mb + j;
                    if (m < cnt) {
                        float v = gelu_tanh(acc[tf][nf][j] + bia);
                        outb[(size_t)(base + m) * Ntot + n] = (bf16)v;
                    }
                }
            } else {
#pragma unroll
                for (int j = 0; j < 8; ++j) {
                    int m = mb + j;
                    if (m < cnt) outf[(size_t)(base + m) * Ntot + n] = acc[tf][nf][j];
                }
            }
        }
    }
}

// ---------------------------------------------------------------- combine
__global__ __launch_bounds__(256) void combine_kernel(
    const float* __restrict__ x, const float* __restrict__ b2,
    const float* __restrict__ ybuf, const int* __restrict__ top_e,
    const float* __restrict__ top_w, const int* __restrict__ slot,
    float* __restrict__ out)
{
    int t = blockIdx.x;
    int e0 = top_e[t * 2], e1 = top_e[t * 2 + 1];
    float w0 = top_w[t * 2], w1v = top_w[t * 2 + 1];
    size_t s0 = (size_t)slot[t * 2], s1 = (size_t)slot[t * 2 + 1];
    const float* xr = x + (size_t)t * HDIM;
    const float* y0 = ybuf + s0 * HDIM;
    const float* y1 = ybuf + s1 * HDIM;
    const float* c0 = b2 + (size_t)e0 * HDIM;
    const float* c1 = b2 + (size_t)e1 * HDIM;
    float* o = out + (size_t)t * HDIM;
#pragma unroll
    for (int i = 0; i < 4; ++i) {
        int h = threadIdx.x + i * 256;
        o[h] = xr[h] + w0 * (y0[h] + c0[h]) + w1v * (y1[h] + c1[h]);
    }
    if (blockIdx.x == 0 && threadIdx.x == 0) {
        out[(size_t)TOK * HDIM + 0] = 0.f;   // act_loss
        out[(size_t)TOK * HDIM + 1] = 0.f;   // ponder_cost
    }
}

// ------------------------------------------------------------------- host
extern "C" void kernel_launch(void* const* d_in, const int* in_sizes, int n_in,
                              void* d_out, int out_size, void* d_ws, size_t ws_size,
                              hipStream_t stream) {
    (void)in_sizes; (void)n_in; (void)out_size; (void)ws_size;
    const float* x   = (const float*)d_in[0];
    const float* rw  = (const float*)d_in[1];
    const float* rb  = (const float*)d_in[2];
    const float* lng = (const float*)d_in[3];
    const float* lnb = (const float*)d_in[4];
    const float* w1  = (const float*)d_in[5];
    const float* b1  = (const float*)d_in[6];
    const float* w2  = (const float*)d_in[7];
    const float* b2  = (const float*)d_in[8];
    float* out = (float*)d_out;

    char* ws = (char*)d_ws;
    size_t off = 0;
    auto carve = [&](size_t bytes) -> char* {
        char* p = ws + off;
        off += (bytes + 255) & ~(size_t)255;
        return p;
    };
    bf16*  xhat   = (bf16*)carve((size_t)TOK * HDIM * 2);
    bf16*  w1t    = (bf16*)carve((size_t)NEXP * FDIM * HDIM * 2);
    bf16*  w2t    = (bf16*)carve((size_t)NEXP * HDIM * FDIM * 2);
    float* b1p    = (float*)carve((size_t)NEXP * FDIM * 4);
    int*   top_e  = (int*)carve((size_t)TOK * 2 * 4);
    float* top_w  = (float*)carve((size_t)TOK * 2 * 4);
    int*   slot   = (int*)carve((size_t)TOK * 2 * 4);
    int*   idx    = (int*)carve((size_t)TOK * 2 * 4);
    int*   cnt    = (int*)carve(NEXP * 4);
    int*   fill   = (int*)carve(NEXP * 4);
    int*   segoff = (int*)carve((NEXP + 1) * 4);
    bf16*  hbuf   = (bf16*)carve((size_t)TOK * TOPK * FDIM * 2);
    float* ybuf   = (float*)carve((size_t)TOK * TOPK * HDIM * 4);

    zero_cnt_kernel<<<1, 32, 0, stream>>>(cnt);
    router_ln_kernel<<<TOK, 256, 0, stream>>>(x, rw, rb, xhat, top_e, top_w, cnt);
    scan_kernel<<<1, 32, 0, stream>>>(cnt, segoff, fill);
    assign_kernel<<<TOK / 256, 256, 0, stream>>>(top_e, segoff, fill, idx, slot);

    tr_w1_kernel<<<dim3(FDIM / 32, HDIM / 32, NEXP), 256, 0, stream>>>(w1, lng, w1t);
    tr_w2_kernel<<<dim3(HDIM / 32, FDIM / 32, NEXP), 256, 0, stream>>>(w2, w2t);
    foldb1_kernel<<<dim3(FDIM / 256, NEXP), 256, 0, stream>>>(w1, lnb, b1, b1p);

    // GEMM1: h = gelu(xhat @ W1' + b1')  [per-expert gathered rows], K=H, N=F
    moe_gemm_kernel<1><<<dim3(FDIM / TN, TOK / TM, NEXP), 256, SMEM_BYTES, stream>>>(
        xhat, HDIM, w1t, FDIM, b1p, idx, segoff, hbuf, nullptr);
    // GEMM2: y = h @ W2ᵀ  [segment-contiguous rows], K=F, N=H
    moe_gemm_kernel<0><<<dim3(HDIM / TN, TOK / TM, NEXP), 256, SMEM_BYTES, stream>>>(
        hbuf, FDIM, w2t, HDIM, nullptr, nullptr, segoff, nullptr, ybuf);

    combine_kernel<<<TOK, 256, 0, stream>>>(x, b2, ybuf, top_e, top_w, slot, out);
}
